// SubGraphGINE_40750649704709
// MI455X (gfx1250) — matrix operand order repaired and checked by reading, not compiled, not run on hardware
//
#include <hip/hip_runtime.h>

typedef __attribute__((ext_vector_type(16))) _Float16 v16h;
typedef __attribute__((ext_vector_type(8)))  _Float16 v8h;
typedef __attribute__((ext_vector_type(2)))  _Float16 h2;
typedef __attribute__((ext_vector_type(8)))  float    v8f;
typedef __attribute__((ext_vector_type(4)))  int      v4i;

#if defined(__AMDGCN__) && __has_builtin(__builtin_amdgcn_global_load_async_to_lds_b128) && __has_builtin(__builtin_amdgcn_s_wait_asynccnt)
#define ASYNC_LDS 1
typedef __attribute__((address_space(1))) v4i* gv4i_p;   // global 128-bit payload ptr
typedef __attribute__((address_space(3))) v4i* lv4i_p;   // LDS 128-bit payload ptr
#else
#define ASYNC_LDS 0
#endif

// ---------------------------------------------------------------------------
// Fragment loads (CDNA5 ISA 7.12.2, wave32). Both are 2x ds_load_b128.
// A tile in LDS: row-major [16 m][32 k].  lane: m=lane&15, hlf=lane>>4
//   frag = concat(As[m][hlf*8 .. +7], As[m][16+hlf*8 .. +7])
// B tile in LDS: column-major [128 n][32 k]. lane: n=lane&15, hlf=lane>>4
//   frag = concat(Bs[n][hlf*8 .. +7], Bs[n][16+hlf*8 .. +7])  (two v8h)
// ---------------------------------------------------------------------------
__device__ __forceinline__ v16h frag_a(const _Float16* As) {
    const int lane = threadIdx.x & 31;
    const _Float16* p = As + (lane & 15) * 32 + (lane >> 4) * 8;
    const v8h a0 = *(const v8h*)p;
    const v8h a1 = *(const v8h*)(p + 16);
    return __builtin_shufflevector(a0, a1, 0,1,2,3,4,5,6,7,8,9,10,11,12,13,14,15);
}
__device__ __forceinline__ v16h frag_b(const _Float16* Bs, int wave) {
    const int lane = threadIdx.x & 31;
    const _Float16* p = Bs + (wave * 16 + (lane & 15)) * 32 + (lane >> 4) * 8;
    const v8h b0 = *(const v8h*)p;
    const v8h b1 = *(const v8h*)(p + 16);
    return __builtin_shufflevector(b0, b1, 0,1,2,3,4,5,6,7,8,9,10,11,12,13,14,15);
}

// copy one packed 32x128 weight tile (4096 f16 = 8KB) global -> LDS
__device__ __forceinline__ void stage_b_tile(const _Float16* __restrict__ src,
                                             _Float16* dst) {
    const int tid = threadIdx.x;
#if ASYNC_LDS
    __builtin_amdgcn_global_load_async_to_lds_b128(
        (gv4i_p)(src + tid * 8), (lv4i_p)(dst + tid * 8), 0, 0);
    __builtin_amdgcn_global_load_async_to_lds_b128(
        (gv4i_p)(src + 2048 + tid * 8), (lv4i_p)(dst + 2048 + tid * 8), 0, 0);
#else
    *(v8h*)(dst + tid * 8)        = *(const v8h*)(src + tid * 8);
    *(v8h*)(dst + 2048 + tid * 8) = *(const v8h*)(src + 2048 + tid * 8);
#endif
}
__device__ __forceinline__ void wait_async() {
#if ASYNC_LDS
    __builtin_amdgcn_s_wait_asynccnt(0);
#endif
}

// ---------------------------------------------------------------------------
// Pre-pack a weight W[K,N] (fp32, row-major) into f16 tiles:
// layout [cb][chunk][n 0..127][k 0..31], zero-padded past K.
// ---------------------------------------------------------------------------
__global__ __launch_bounds__(256) void pack_weight_kernel(
    const float* __restrict__ W, _Float16* __restrict__ Wp,
    int K, int N, int total)
{
    const int e = blockIdx.x * 256 + threadIdx.x;
    if (e >= total) return;
    const int nChunks = (K + 31) >> 5;
    const int inTile = e & 4095;
    const int tile   = e >> 12;
    const int c  = tile % nChunks;
    const int cb = tile / nChunks;
    const int n  = inTile >> 5;
    const int k  = inTile & 31;
    const int gk = c * 32 + k;
    const float v = (gk < K) ? W[(size_t)gk * N + cb * 128 + n] : 0.f;
    Wp[e] = (_Float16)v;
}

// ---------------------------------------------------------------------------
// Node GEMM: C[M,N] = f(A) @ W + bias
//   MODE 0: f(A)=A      MODE 1: f(A)=A+A2      MODE 2: f(A)=relu(A*scale+shift)
// Block: 256 thr = 8 waves, tile 16(M) x 128(N), K chunked by 32,
// double-buffered LDS, async weight-tile copy.
// ---------------------------------------------------------------------------
template <int MODE>
__global__ __launch_bounds__(256) void gemm_node_kernel(
    const float* __restrict__ A, const float* __restrict__ A2,
    const _Float16* __restrict__ Wp, const float* __restrict__ bias,
    float* __restrict__ C, int M, int K, int N,
    const float* __restrict__ inScale, const float* __restrict__ inShift)
{
    __shared__ __align__(16) _Float16 As[2][16 * 32];
    __shared__ __align__(16) _Float16 Bs[2][32 * 128];

    const int tid   = threadIdx.x;
    const int wave  = tid >> 5;
    const int lane  = tid & 31;
    const int hlf   = lane >> 4;
    const int ln    = lane & 15;
    const int mBase = blockIdx.x * 16;
    const int nChunks = (K + 31) >> 5;
    const _Float16* wCol = Wp + (size_t)blockIdx.y * nChunks * 4096;

    auto stageA = [&](int c, int buf) {
        const int e = tid * 2;                 // 512 f16 / 256 threads
        const int r = e >> 5, kk = e & 31;
        const int m = mBase + r;
        const int k = (c << 5) + kk;
        float v0 = 0.f, v1 = 0.f;
        if (m < M) {
            if (k < K) {
                v0 = A[(size_t)m * K + k];
                if constexpr (MODE == 1) v0 += A2[(size_t)m * K + k];
                if constexpr (MODE == 2) v0 = fmaxf(fmaf(v0, inScale[k], inShift[k]), 0.f);
            }
            if (k + 1 < K) {
                v1 = A[(size_t)m * K + k + 1];
                if constexpr (MODE == 1) v1 += A2[(size_t)m * K + k + 1];
                if constexpr (MODE == 2) v1 = fmaxf(fmaf(v1, inScale[k + 1], inShift[k + 1]), 0.f);
            }
        }
        h2 p; p[0] = (_Float16)v0; p[1] = (_Float16)v1;
        *(h2*)&As[buf][r * 32 + kk] = p;
    };

    v8f acc = {};
    stageA(0, 0);
    stage_b_tile(wCol, Bs[0]);

    for (int c = 0; c < nChunks; ++c) {
        const int buf = c & 1;
        wait_async();
        __syncthreads();
        if (c + 1 < nChunks) {
            stageA(c + 1, buf ^ 1);
            stage_b_tile(wCol + (size_t)(c + 1) * 4096, Bs[buf ^ 1]);
        }
        const v16h a = frag_a(As[buf]);
        const v16h b = frag_b(Bs[buf], wave);
        acc = __builtin_amdgcn_wmma_f32_16x16x32_f16(
            false, a, false, b, (short)0, acc, false, false);
    }

    const int n   = blockIdx.y * 128 + wave * 16 + ln;
    const float bv = bias[n];
#pragma unroll
    for (int i = 0; i < 8; ++i) {
        const int m = mBase + i + 8 * hlf;
        if (m < M) C[(size_t)m * N + n] = acc[i] + bv;
    }
}

// ---------------------------------------------------------------------------
// Edge GEMM with fused message epilogue:
//   acc = edge_attr @ We ; val = relu(acc + be[n] + h[src[m]][n])
//   atomicAdd(agg[dst[m]][n], val)   (agg 51MB -> L2 resident)
// ---------------------------------------------------------------------------
__global__ __launch_bounds__(256) void edge_gemm_scatter_kernel(
    const float* __restrict__ EA, const _Float16* __restrict__ Wp,
    const float* __restrict__ be, const float* __restrict__ h,
    const int* __restrict__ src, const int* __restrict__ dst,
    float* __restrict__ agg, int Ee, int K)
{
    __shared__ __align__(16) _Float16 As[2][16 * 32];
    __shared__ __align__(16) _Float16 Bs[2][32 * 128];

    const int tid   = threadIdx.x;
    const int wave  = tid >> 5;
    const int lane  = tid & 31;
    const int hlf   = lane >> 4;
    const int ln    = lane & 15;
    const int mBase = blockIdx.x * 16;
    const int nChunks = (K + 31) >> 5;

    auto stageA = [&](int c, int buf) {
        const int e = tid * 2;
        const int r = e >> 5, kk = e & 31;
        const int m = mBase + r;
        const int k = (c << 5) + kk;
        float v0 = 0.f, v1 = 0.f;
        if (m < Ee) {
            if (k < K)     v0 = EA[(size_t)m * K + k];
            if (k + 1 < K) v1 = EA[(size_t)m * K + k + 1];
        }
        h2 p; p[0] = (_Float16)v0; p[1] = (_Float16)v1;
        *(h2*)&As[buf][r * 32 + kk] = p;
    };

    v8f acc = {};
    stageA(0, 0);
    stage_b_tile(Wp, Bs[0]);

    for (int c = 0; c < nChunks; ++c) {
        const int buf = c & 1;
        wait_async();
        __syncthreads();
        if (c + 1 < nChunks) {
            stageA(c + 1, buf ^ 1);
            stage_b_tile(Wp + (size_t)(c + 1) * 4096, Bs[buf ^ 1]);
        }
        const v16h a = frag_a(As[buf]);
        const v16h b = frag_b(Bs[buf], wave);
        acc = __builtin_amdgcn_wmma_f32_16x16x32_f16(
            false, a, false, b, (short)0, acc, false, false);
    }

    const int n   = wave * 16 + ln;
    const float bv = be[n];
#pragma unroll
    for (int i = 0; i < 8; ++i) {
        const int m = mBase + i + 8 * hlf;       // edge id
        if (m < Ee) {
            const int s = src[m], d = dst[m];
            float val = acc[i] + bv + h[(size_t)s * 128 + n];
            val = fmaxf(val, 0.f);
            atomicAdd(&agg[(size_t)d * 128 + n], val);
        }
    }
}

// ---------------------------------------------------------------------------
// BatchNorm helpers
// ---------------------------------------------------------------------------
__global__ __launch_bounds__(256) void colstats_kernel(
    const float* __restrict__ Y, int M, int N,
    float* __restrict__ sum, float* __restrict__ sumsq)
{
    const int perRow  = 256 / N;
    const int col     = threadIdx.x % N;
    const int rstart  = blockIdx.x * perRow + threadIdx.x / N;
    const int rstride = gridDim.x * perRow;
    float s = 0.f, q = 0.f;
    for (int m = rstart; m < M; m += rstride) {
        const float v = Y[(size_t)m * N + col];
        s += v; q += v * v;
    }
    atomicAdd(&sum[col], s);
    atomicAdd(&sumsq[col], q);
}

__global__ void bn_finalize_kernel(
    const float* __restrict__ sum, const float* __restrict__ sumsq,
    const float* __restrict__ gamma, const float* __restrict__ beta,
    float* __restrict__ scale, float* __restrict__ shift, int N, float invM)
{
    const int n = blockIdx.x * blockDim.x + threadIdx.x;
    if (n < N) {
        const float mean = sum[n] * invM;
        const float var  = sumsq[n] * invM - mean * mean;
        const float sc   = gamma[n] * rsqrtf(var + 1e-5f);
        scale[n] = sc;
        shift[n] = beta[n] - mean * sc;
    }
}

__global__ void bn_apply_kernel(
    const float* __restrict__ Y, float* __restrict__ H,
    const float* __restrict__ scale, const float* __restrict__ shift,
    long long total, int N, int relu)
{
    const long long i = (long long)blockIdx.x * blockDim.x + threadIdx.x;
    if (i < total) {
        const int n = (int)(i % N);
        float v = fmaf(Y[i], scale[n], shift[n]);
        if (relu) v = fmaxf(v, 0.f);
        H[i] = v;
    }
}

// ---------------------------------------------------------------------------
// Global mean pool
// ---------------------------------------------------------------------------
__global__ void pool_acc_kernel(
    const float* __restrict__ h, const int* __restrict__ batch,
    float* __restrict__ gsum, float* __restrict__ gcnt, long long total)
{
    const long long i = (long long)blockIdx.x * blockDim.x + threadIdx.x;
    if (i < total) {
        const int m = (int)(i >> 7);
        const int n = (int)(i & 127);
        const int g = batch[m];
        atomicAdd(&gsum[(size_t)g * 128 + n], h[i]);
        if (n == 0) atomicAdd(&gcnt[g], 1.0f);
    }
}

__global__ void pool_div_kernel(
    const float* __restrict__ gsum, const float* __restrict__ gcnt,
    float* __restrict__ out, int total)
{
    const int i = blockIdx.x * blockDim.x + threadIdx.x;
    if (i < total) {
        const int g = i >> 7;
        out[i] = gsum[i] / fmaxf(gcnt[g], 1.0f);
    }
}

// ---------------------------------------------------------------------------
extern "C" void kernel_launch(void* const* d_in, const int* in_sizes, int n_in,
                              void* d_out, int out_size, void* d_ws, size_t ws_size,
                              hipStream_t stream)
{
    const float* x     = (const float*)d_in[0];
    const float* ea    = (const float*)d_in[1];
    const int*   src   = (const int*)d_in[2];
    const int*   dst   = (const int*)d_in[3];
    const int*   batch = (const int*)d_in[4];
    const float* W_emb = (const float*)d_in[5];
    const float* b_emb = (const float*)d_in[6];

    const int Nn  = in_sizes[4];          // nodes
    const int Ee  = in_sizes[2];          // edges
    const int Din = in_sizes[0] / Nn;     // 73
    const int De  = in_sizes[1] / Ee;     // 101
    const int D   = in_sizes[6];          // 128
    const int D2  = in_sizes[10];         // 256
    const int G   = out_size / D;         // graphs

    // ---- workspace carve-up ----
    float* hA     = (float*)d_ws;
    float* hB     = hA + (size_t)Nn * D;
    float* agg    = hB + (size_t)Nn * D;
    float* y1     = agg + (size_t)Nn * D;
    float* y2     = y1 + (size_t)Nn * D2;
    float* colsum = y2 + (size_t)Nn * D;
    float* colsq  = colsum + 256;
    float* scale  = colsq + 256;
    float* shift  = scale + 256;
    float* gsum   = shift + 256;
    float* gcnt   = gsum + (size_t)G * D;
    // f16 packed-weight region (16B aligned)
    uintptr_t wpAddr = ((uintptr_t)(gcnt + G) + 15) & ~(uintptr_t)15;
    _Float16* wp = (_Float16*)wpAddr;

    auto packedSize = [](int K, int N) -> size_t {
        return (size_t)((K + 31) / 32) * (N / 128) * 4096;
    };
    _Float16* wpEmb = wp;
    _Float16* wpL[2][3];   // [layer][We,W1,W2]
    {
        _Float16* p = wpEmb + packedSize(Din, D);
        for (int i = 0; i < 2; ++i) {
            wpL[i][0] = p; p += packedSize(De, D);
            wpL[i][1] = p; p += packedSize(D, D2);
            wpL[i][2] = p; p += packedSize(D2, D);
        }
    }

    const dim3 blk(256);
    auto pack = [&](const float* W, _Float16* Wp, int K, int N) {
        const int total = (int)packedSize(K, N);
        pack_weight_kernel<<<(total + 255) / 256, blk, 0, stream>>>(W, Wp, K, N, total);
    };
    pack(W_emb, wpEmb, Din, D);
    for (int i = 0; i < 2; ++i) {
        pack((const float*)d_in[7 + 10 * i + 0], wpL[i][0], De, D);
        pack((const float*)d_in[7 + 10 * i + 2], wpL[i][1], D, D2);
        pack((const float*)d_in[7 + 10 * i + 6], wpL[i][2], D2, D);
    }

    const int mBlk = (Nn + 15) / 16;
    const float invM = 1.0f / (float)Nn;

    // h = x @ W_emb + b_emb
    float* hcur = hA; float* hnext = hB;
    gemm_node_kernel<0><<<dim3(mBlk, D / 128), blk, 0, stream>>>(
        x, nullptr, wpEmb, b_emb, hcur, Nn, Din, D, nullptr, nullptr);

    for (int i = 0; i < 2; ++i) {
        const float* be  = (const float*)d_in[7 + 10 * i + 1];
        const float* b1  = (const float*)d_in[7 + 10 * i + 3];
        const float* g1  = (const float*)d_in[7 + 10 * i + 4];
        const float* bb1 = (const float*)d_in[7 + 10 * i + 5];
        const float* b2  = (const float*)d_in[7 + 10 * i + 7];
        const float* go  = (const float*)d_in[7 + 10 * i + 8];
        const float* bo  = (const float*)d_in[7 + 10 * i + 9];

        // agg = scatter_sum(relu(h[src] + edge_attr@We + be), dst)
        (void)hipMemsetAsync(agg, 0, (size_t)Nn * D * sizeof(float), stream);
        edge_gemm_scatter_kernel<<<dim3((Ee + 15) / 16), blk, 0, stream>>>(
            ea, wpL[i][0], be, hcur, src, dst, agg, Ee, De);

        // y1 = (h + agg) @ W1 + b1
        gemm_node_kernel<1><<<dim3(mBlk, D2 / 128), blk, 0, stream>>>(
            hcur, agg, wpL[i][1], b1, y1, Nn, D, D2, nullptr, nullptr);

        // BN stats y1 -> scale/shift (g1, bb1)
        (void)hipMemsetAsync(colsum, 0, 512 * sizeof(float), stream);
        colstats_kernel<<<512, blk, 0, stream>>>(y1, Nn, D2, colsum, colsq);
        bn_finalize_kernel<<<1, 256, 0, stream>>>(colsum, colsq, g1, bb1,
                                                  scale, shift, D2, invM);

        // y2 = relu(BN(y1)) @ W2 + b2 (BN+ReLU fused into A stage)
        gemm_node_kernel<2><<<dim3(mBlk, D / 128), blk, 0, stream>>>(
            y1, nullptr, wpL[i][2], b2, y2, Nn, D2, D, scale, shift);

        // BN stats y2 -> scale/shift (go, bo); h_next = BN(y2) (+ReLU except last)
        (void)hipMemsetAsync(colsum, 0, 512 * sizeof(float), stream);
        colstats_kernel<<<512, blk, 0, stream>>>(y2, Nn, D, colsum, colsq);
        bn_finalize_kernel<<<1, 256, 0, stream>>>(colsum, colsq, go, bo,
                                                  scale, shift, D, invM);
        const long long tot = (long long)Nn * D;
        bn_apply_kernel<<<(unsigned)((tot + 255) / 256), blk, 0, stream>>>(
            y2, hnext, scale, shift, tot, D, (i == 0) ? 1 : 0);

        float* t = hcur; hcur = hnext; hnext = t;
    }

    // global mean pool
    (void)hipMemsetAsync(gsum, 0, ((size_t)G * D + G) * sizeof(float), stream);
    const long long tot = (long long)Nn * D;
    pool_acc_kernel<<<(unsigned)((tot + 255) / 256), blk, 0, stream>>>(
        hcur, batch, gsum, gcnt, tot);
    pool_div_kernel<<<(G * D + 255) / 256, blk, 0, stream>>>(
        gsum, gcnt, (float*)d_out, G * D);
}
